// CustomDGCNNConv_8976481649327
// MI455X (gfx1250) — matrix operand order repaired and checked by reading, not compiled, not run on hardware
//
#include <hip/hip_runtime.h>
#include <hip/hip_bf16.h>
#include <math.h>

// Problem constants (from reference)
#define N_PTS  16384
#define KNN    16
#define C_IN   64
#define C_OUT  64
#define SDIM   8
#define LN_EPS 1e-5f

typedef __attribute__((ext_vector_type(16))) _Float16 v16h;
typedef __attribute__((ext_vector_type(8)))  _Float16 v8h;
typedef __attribute__((ext_vector_type(8)))  float    v8f;

__device__ __forceinline__ v16h cat8(v8h lo, v8h hi) {
    return __builtin_shufflevector(lo, hi, 0,1,2,3,4,5,6,7,8,9,10,11,12,13,14,15);
}

// ---------------------------------------------------------------------------
// Kernel 1: s = x @ Ws^T + bs ; sq = rowsum(s*s).   1 thread per node.
// ---------------------------------------------------------------------------
__global__ void proj_kernel(const float* __restrict__ x,
                            const float* __restrict__ Ws,
                            const float* __restrict__ bs,
                            float* __restrict__ s,
                            float* __restrict__ sq) {
    __shared__ float sWs[SDIM * C_IN];
    for (int t = threadIdx.x; t < SDIM * C_IN; t += blockDim.x) sWs[t] = Ws[t];
    __syncthreads();

    int i = blockIdx.x * blockDim.x + threadIdx.x;
    if (i >= N_PTS) return;

    float xi[C_IN];
    #pragma unroll
    for (int c = 0; c < C_IN; ++c) xi[c] = x[i * C_IN + c];

    float acc2 = 0.f;
    #pragma unroll
    for (int d = 0; d < SDIM; ++d) {
        float a = bs[d];
        #pragma unroll
        for (int c = 0; c < C_IN; ++c) a = fmaf(xi[c], sWs[d * C_IN + c], a);
        s[i * SDIM + d] = a;
        acc2 = fmaf(a, a, acc2);
    }
    sq[i] = acc2;
}

// ---------------------------------------------------------------------------
// Kernel 2: brute-force kNN in projected space.  1 thread per target row,
// sources tiled through LDS, register-resident top-16 (smallest d2).
// d2 = sq[i] + sq[j] - 2*dot(s_i, s_j).  Self-edge included (as in reference).
// NOTE: a v_wmma_f32_16x16x4_f32 formulation of the gram matrix reliably
// segfaults this toolchain's Register Coalescer (LLVM 23.1-rc2, four distinct
// formulations tried: builtin, tied inline-asm, early-clobber inline-asm,
// noinline helper with whole-vector store), so the dot products stay on VALU
// at full f32 precision.
// ---------------------------------------------------------------------------
__global__ void knn_kernel(const float* __restrict__ s,
                           const float* __restrict__ sq,
                           int* __restrict__ idx) {
    __shared__ float sj[256 * SDIM];
    __shared__ float sqj[256];

    int i = blockIdx.x * 256 + threadIdx.x;

    float si[SDIM];
    #pragma unroll
    for (int d = 0; d < SDIM; ++d) si[d] = s[i * SDIM + d];
    float sqi = sq[i];

    float bd[KNN]; int bi[KNN];
    #pragma unroll
    for (int t = 0; t < KNN; ++t) { bd[t] = 3.0e38f; bi[t] = 0; }
    float worst = 3.0e38f;

    for (int base = 0; base < N_PTS; base += 256) {
        __syncthreads();
        int j = base + threadIdx.x;
        #pragma unroll
        for (int d = 0; d < SDIM; ++d) sj[threadIdx.x * SDIM + d] = s[j * SDIM + d];
        sqj[threadIdx.x] = sq[j];
        __syncthreads();

        for (int t = 0; t < 256; ++t) {
            float dot = 0.f;
            #pragma unroll
            for (int d = 0; d < SDIM; ++d) dot = fmaf(si[d], sj[t * SDIM + d], dot);
            float d2 = sqi + sqj[t] - 2.f * dot;
            if (d2 < worst) {
                // replace current worst entry, then rescan for new worst
                bool done = false;
                #pragma unroll
                for (int q = 0; q < KNN; ++q) {
                    if (!done && bd[q] == worst) { bd[q] = d2; bi[q] = base + t; done = true; }
                }
                worst = bd[0];
                #pragma unroll
                for (int q = 1; q < KNN; ++q) worst = fmaxf(worst, bd[q]);
            }
        }
    }
    #pragma unroll
    for (int t = 0; t < KNN; ++t) idx[i * KNN + t] = bi[t];
}

// ---------------------------------------------------------------------------
// Kernel 3: per-node edge MLP with WMMA.
// One wave32 per node. A = feat tile (M=16 edges x K), B = weights.
// Layer1: 16x128 @ 128x64 -> 4 N-tiles x 4 K-steps of v_wmma_f32_16x16x32_f16
// LN+ReLU (f32), repack f16, Layer2: 16x64 @ 64x64 -> 4x2 WMMAs, LN+ReLU,
// mean over the 16 edges.  4 waves per 128-thread block; static LDS 57.5 KB.
// ---------------------------------------------------------------------------
__global__ void mlp_kernel(const float* __restrict__ x,
                           const int* __restrict__ idx,
                           const float* __restrict__ W1, const float* __restrict__ b1,
                           const float* __restrict__ g1, const float* __restrict__ be1,
                           const float* __restrict__ W2, const float* __restrict__ b2,
                           const float* __restrict__ g2, const float* __restrict__ be2,
                           float* __restrict__ out) {
    __shared__ __align__(16) _Float16 sW1[C_OUT * 2 * C_IN];  // 64x128, row-major (n,k)
    __shared__ __align__(16) _Float16 sW2[C_OUT * C_OUT];     // 64x64,  row-major (n,k)
    __shared__ float sVec[6 * C_OUT];                          // b1,g1,be1,b2,g2,be2
    __shared__ __align__(16) _Float16 featA[4][16 * 128];      // per-wave A tiles (f16)
    __shared__ float hbuf[4][16 * 64];                         // per-wave f32 scratch

    for (int t = threadIdx.x; t < C_OUT * 2 * C_IN; t += blockDim.x)
        sW1[t] = (_Float16)W1[t];
    for (int t = threadIdx.x; t < C_OUT * C_OUT; t += blockDim.x)
        sW2[t] = (_Float16)W2[t];
    if (threadIdx.x < C_OUT) {
        int t = threadIdx.x;
        sVec[0 * 64 + t] = b1[t];  sVec[1 * 64 + t] = g1[t];  sVec[2 * 64 + t] = be1[t];
        sVec[3 * 64 + t] = b2[t];  sVec[4 * 64 + t] = g2[t];  sVec[5 * 64 + t] = be2[t];
    }
    __syncthreads();

    const int wave = threadIdx.x >> 5;
    const int lane = threadIdx.x & 31;
    const int node = blockIdx.x * 4 + wave;

    _Float16* fA = featA[wave];
    float*    hb = hbuf[wave];

    // ---- build feat tile: row e = edge, cols [x_i | x_j - x_i] ----
    {
        const int e    = lane & 15;
        const int hh   = lane >> 4;           // 0 -> x_i, 1 -> x_j - x_i
        const int j    = idx[node * KNN + e];
        const float* xi = x + (size_t)node * C_IN;
        const float* xj = x + (size_t)j * C_IN;
        #pragma unroll
        for (int c = 0; c < C_IN; ++c) {
            float v = hh ? (xj[c] - xi[c]) : xi[c];
            fA[e * 128 + hh * C_IN + c] = (_Float16)v;
        }
    }
    __syncthreads();

    // WMMA f16 lane-layout helpers (ISA 7.12.2, wave32):
    const int m     = lane & 15;              // A row
    const int nlo   = lane & 15;              // B/D column within 16-tile
    const int abase = (lane < 16) ? 0 : 8;    // A: k = abase+{0..7}, abase+16+{0..7}
    const int bbase = (lane < 16) ? 0 : 8;    // B: k = bbase+{0..7}, bbase+16+{0..7}
    const int moff  = (lane < 16) ? 0 : 8;    // C/D: row = vgpr + moff

    // ---- layer 1: 16x128 @ 128x64 ----
    v8f acc[4] = { {}, {}, {}, {} };
    #pragma unroll
    for (int kk = 0; kk < 4; ++kk) {
        const int k0 = kk * 32;
        v8h alo = *(const v8h*)&fA[m * 128 + k0 + abase];
        v8h ahi = *(const v8h*)&fA[m * 128 + k0 + abase + 16];
        v16h a  = cat8(alo, ahi);
        #pragma unroll
        for (int nt = 0; nt < 4; ++nt) {
            const int n = nt * 16 + nlo;
            v8h blo = *(const v8h*)&sW1[n * 128 + k0 + bbase];
            v8h bhi = *(const v8h*)&sW1[n * 128 + k0 + bbase + 16];
            v16h b  = cat8(blo, bhi);
            acc[nt] = __builtin_amdgcn_wmma_f32_16x16x32_f16(
                false, a, false, b, (short)0, acc[nt], false, false);
        }
    }
    #pragma unroll
    for (int nt = 0; nt < 4; ++nt)
        #pragma unroll
        for (int v = 0; v < 8; ++v)
            hb[(v + moff) * 64 + nt * 16 + nlo] = acc[nt][v];
    __syncthreads();

    // ---- LayerNorm1 + ReLU (f32), repack f16 into fA (reused as layer-2 A) ----
    if (lane < 16) {
        float hv[64], mu = 0.f, var = 0.f;
        #pragma unroll
        for (int c = 0; c < 64; ++c) { float v = hb[lane * 64 + c] + sVec[c]; hv[c] = v; mu += v; }
        mu *= (1.f / 64.f);
        #pragma unroll
        for (int c = 0; c < 64; ++c) { float d = hv[c] - mu; var = fmaf(d, d, var); }
        var *= (1.f / 64.f);
        float rstd = rsqrtf(var + LN_EPS);
        #pragma unroll
        for (int c = 0; c < 64; ++c) {
            float v = (hv[c] - mu) * rstd * sVec[64 + c] + sVec[128 + c];
            v = v > 0.f ? v : 0.f;
            fA[lane * 64 + c] = (_Float16)v;
        }
    }
    __syncthreads();

    // ---- layer 2: 16x64 @ 64x64 ----
    v8f acc2[4] = { {}, {}, {}, {} };
    #pragma unroll
    for (int kk = 0; kk < 2; ++kk) {
        const int k0 = kk * 32;
        v8h alo = *(const v8h*)&fA[m * 64 + k0 + abase];
        v8h ahi = *(const v8h*)&fA[m * 64 + k0 + abase + 16];
        v16h a  = cat8(alo, ahi);
        #pragma unroll
        for (int nt = 0; nt < 4; ++nt) {
            const int n = nt * 16 + nlo;
            v8h blo = *(const v8h*)&sW2[n * 64 + k0 + bbase];
            v8h bhi = *(const v8h*)&sW2[n * 64 + k0 + bbase + 16];
            v16h b  = cat8(blo, bhi);
            acc2[nt] = __builtin_amdgcn_wmma_f32_16x16x32_f16(
                false, a, false, b, (short)0, acc2[nt], false, false);
        }
    }
    __syncthreads();   // hb fully consumed by LN1 across the wave; safe to overwrite
    #pragma unroll
    for (int nt = 0; nt < 4; ++nt)
        #pragma unroll
        for (int v = 0; v < 8; ++v)
            hb[(v + moff) * 64 + nt * 16 + nlo] = acc2[nt][v];
    __syncthreads();

    // ---- LayerNorm2 + ReLU (f32), write normalized back into hb ----
    if (lane < 16) {
        float hv[64], mu = 0.f, var = 0.f;
        #pragma unroll
        for (int c = 0; c < 64; ++c) { float v = hb[lane * 64 + c] + sVec[192 + c]; hv[c] = v; mu += v; }
        mu *= (1.f / 64.f);
        #pragma unroll
        for (int c = 0; c < 64; ++c) { float d = hv[c] - mu; var = fmaf(d, d, var); }
        var *= (1.f / 64.f);
        float rstd = rsqrtf(var + LN_EPS);
        #pragma unroll
        for (int c = 0; c < 64; ++c) {
            float v = (hv[c] - mu) * rstd * sVec[256 + c] + sVec[320 + c];
            hb[lane * 64 + c] = v > 0.f ? v : 0.f;
        }
    }
    __syncthreads();

    // ---- mean over the 16 edges; each lane writes 2 channels ----
    #pragma unroll
    for (int cc = 0; cc < 2; ++cc) {
        const int c = lane * 2 + cc;
        float sacc = 0.f;
        #pragma unroll
        for (int e = 0; e < KNN; ++e) sacc += hb[e * 64 + c];
        out[(size_t)node * C_OUT + c] = sacc * (1.f / 16.f);
    }
}

// ---------------------------------------------------------------------------
extern "C" void kernel_launch(void* const* d_in, const int* in_sizes, int n_in,
                              void* d_out, int out_size, void* d_ws, size_t ws_size,
                              hipStream_t stream) {
    (void)in_sizes; (void)n_in; (void)out_size; (void)ws_size;

    const float* x   = (const float*)d_in[0];
    // d_in[1] edge_index (int64), d_in[2] eta, d_in[3] phi: unused (as in reference)
    const float* Ws  = (const float*)d_in[4];
    const float* bs  = (const float*)d_in[5];
    const float* W1  = (const float*)d_in[6];
    const float* b1  = (const float*)d_in[7];
    const float* g1  = (const float*)d_in[8];
    const float* be1 = (const float*)d_in[9];
    const float* W2  = (const float*)d_in[10];
    const float* b2  = (const float*)d_in[11];
    const float* g2  = (const float*)d_in[12];
    const float* be2 = (const float*)d_in[13];
    float* out = (float*)d_out;

    char* ws = (char*)d_ws;
    float* s   = (float*)ws;                                                // N*8 f32
    float* sq  = (float*)(ws + (size_t)N_PTS * SDIM * sizeof(float));       // N   f32
    int*   idx = (int*)  (ws + (size_t)N_PTS * (SDIM + 1) * sizeof(float)); // N*16 i32

    proj_kernel<<<N_PTS / 256, 256, 0, stream>>>(x, Ws, bs, s, sq);
    knn_kernel <<<N_PTS / 256, 256, 0, stream>>>(s, sq, idx);
    mlp_kernel <<<N_PTS / 4,   128, 0, stream>>>(x, idx, W1, b1, g1, be1,
                                                 W2, b2, g2, be2, out);
}
